// Predict_72284299592166
// MI455X (gfx1250) — compile-verified, matrix operand order checked by reading, not run on hardware
//
#include <hip/hip_runtime.h>

// Problem constants (match reference)
#define Bb   512   // batch
#define Tt   32    // time steps
#define INx  8     // input features
#define IN1  9     // input + y_prev
#define Hh   64    // hidden
#define Pp   256   // particles
#define G4   256   // 4*H gates

typedef __attribute__((ext_vector_type(16))) __bf16 v16bf;
typedef __attribute__((ext_vector_type(8)))  float  v8f;

union BFV { v16bf v; unsigned short u[16]; };
union FU  { float f; unsigned u; };

__device__ __forceinline__ float bf2f(unsigned short h) {
    FU x; x.u = ((unsigned)h) << 16; return x.f;
}
__device__ __forceinline__ unsigned short f2bf(float f) {
    FU x; x.f = f;
    unsigned u = x.u + 0x7FFFu + ((x.u >> 16) & 1u);  // round-to-nearest-even
    return (unsigned short)(u >> 16);
}
__device__ __forceinline__ float sigm(float x) { return 1.0f / (1.0f + expf(-x)); }

__device__ __forceinline__ unsigned pcg(unsigned v) {
    v = v * 747796405u + 2891336453u;
    unsigned w = ((v >> ((v >> 28) + 4u)) ^ v) * 277803737u;
    return (w >> 22) ^ w;
}
__device__ __forceinline__ float gaussian(unsigned seed) {
    unsigned a = pcg(seed), b = pcg(seed ^ 0x6E624EB7u);
    float u1 = (float)(a + 1u) * 2.3283064e-10f;   // (0,1]
    float u2 = (float)b * 2.3283064e-10f;
    return sqrtf(-2.0f * logf(u1)) * cosf(6.28318531f * u2);
}

// Load a 16x32 bf16 fragment in CDNA5 WMMA A/B register layout from an LDS
// row-major [rows][64] bf16 tile.  lane&15 selects row (A: M, B: N);
// per-lane element e=2v+c maps to K = (v<4 ? 2v : 8+2v) + c (+8 for hi lanes).
__device__ __forceinline__ v16bf load_frag(const unsigned short* base, int row0,
                                           int kbase, int lane) {
    const int r  = lane & 15;
    const int hi = (lane >> 4) & 1;
    const unsigned short* p = base + (row0 + r) * Hh + kbase + hi * 8;
    BFV f;
#pragma unroll
    for (int v = 0; v < 8; ++v) {
        int K = (v < 4) ? (2 * v) : (8 + 2 * v);
        f.u[2 * v]     = p[K];
        f.u[2 * v + 1] = p[K + 1];
    }
    return f.v;
}

// ---------------------------------------------------------------- init / convert
__global__ void init_kernel(unsigned short* __restrict__ h_bf,
                            float* __restrict__ c_st, float* __restrict__ wacc) {
    size_t i = (size_t)blockIdx.x * blockDim.x + threadIdx.x;
    const size_t N = (size_t)Pp * Bb * Hh;
    for (size_t k = i; k < N; k += (size_t)gridDim.x * blockDim.x) {
        h_bf[k] = 0;
        c_st[k] = 0.0f;
    }
    if (i < Bb) wacc[i] = 0.0f;
}

__global__ void conv_whh_kernel(const float* __restrict__ W_hh,
                                unsigned short* __restrict__ whh_bf) {
    int i = blockIdx.x * blockDim.x + threadIdx.x;
    if (i < G4 * Hh) whh_bf[i] = f2bf(W_hh[i]);
}

// ---------------------------------------------------------------- fused LSTM step
// One block per batch element, 512 threads = 16 waves (wave32).
// Fuses: hidden-mean over particles, softplus std, x@W_ih.T + biases,
// WMMA bf16 GEMM gates[256,256] = h[256,64] @ W_hh.T (+xg), LSTM elementwise,
// reparam noise, bitonic argsort over particles, sorted write-back + logpdf.
__global__ __launch_bounds__(512) void lstm_step_kernel(
    const float* __restrict__ input_data, const float* __restrict__ y_prev,
    const float* __restrict__ Wv, const float* __restrict__ bv,
    const float* __restrict__ W_ih, const float* __restrict__ b_ih,
    const float* __restrict__ b_hh, const float* __restrict__ Wp,
    const float* __restrict__ bp, const float* __restrict__ Wf,
    const float* __restrict__ bff, unsigned short* __restrict__ h_bf,
    float* __restrict__ c_st, const unsigned short* __restrict__ whh_bf,
    float* __restrict__ wacc, int t) {
    extern __shared__ char smem[];
    unsigned short* hA    = (unsigned short*)smem;             // 256*64 bf16 (A)
    unsigned short* Wb    = hA + Pp * Hh;                      // 256*64 bf16 (B)
    float*          hrep  = (float*)(Wb + G4 * Hh);            // 256*64 f32
    float*          xg_l  = hrep + Pp * Hh;                    // 256
    float*          std_l = xg_l + G4;                         // 64
    float*          keys  = std_l + Hh;                        // 256
    int*            order = (int*)(keys + Pp);                 // 256
    float*          red   = (float*)(order + Pp);              // 256
    float*          xc    = red + Pp;                          // 16
    float*          mean_l = xc + 16;                          // 64
    float*          partial = mean_l + Hh;                     // 8*64

    const int b = blockIdx.x;
    const int tid = threadIdx.x;
    const int lane = tid & 31;
    const int wave = tid >> 5;

    // ---- stage h tile (bf16 A) + W_hh (bf16 B) + x_cat into LDS ----
    for (int i = tid; i < Pp * Hh; i += 512) {
        int p = i >> 6, hd = i & 63;
        hA[i] = h_bf[(p * Bb + b) * Hh + hd];
        Wb[i] = whh_bf[i];
    }
    if (tid < IN1)
        xc[tid] = (tid < INx) ? input_data[(b * Tt + t) * INx + tid]
                              : y_prev[b * Tt + t];
    __syncthreads();

    // ---- hidden mean over particles (from LDS, 2-level reduction) ----
    {
        const int hd = tid & 63, chunk = tid >> 6;  // 8 chunks of 32 particles
        float s = 0.0f;
        for (int p = chunk * 32; p < chunk * 32 + 32; ++p)
            s += bf2f(hA[p * Hh + hd]);
        partial[chunk * Hh + hd] = s;
    }
    __syncthreads();
    if (tid < Hh) {
        float s = 0.0f;
#pragma unroll
        for (int ch = 0; ch < 8; ++ch) s += partial[ch * Hh + tid];
        mean_l[tid] = s * (1.0f / (float)Pp);
    }
    __syncthreads();

    // ---- std = softplus([x, mean] @ Wv.T + bv);  xg = x@W_ih.T + b_ih + b_hh
    if (tid < Hh) {
        float v = bv[tid];
        const float* w = Wv + tid * (Hh + IN1);
        for (int k = 0; k < IN1; ++k) v += w[k] * xc[k];
        for (int k = 0; k < Hh; ++k) v += w[IN1 + k] * mean_l[k];
        std_l[tid] = (v > 20.0f) ? v : log1pf(expf(v));  // softplus
    }
    if (tid < G4) {
        float s = b_ih[tid] + b_hh[tid];
        const float* w = W_ih + tid * IN1;
        for (int k = 0; k < IN1; ++k) s += w[k] * xc[k];
        xg_l[tid] = s;
    }
    __syncthreads();

    // ---- WMMA strip: 16 particles x all 4 gates, fused elementwise ----
    const int m0  = wave * 16;
    const int hi  = (lane >> 4) & 1;
    const int col = lane & 15;
    v16bf a0 = load_frag(hA, m0, 0, lane);
    v16bf a1 = load_frag(hA, m0, 32, lane);

#pragma unroll
    for (int j = 0; j < 4; ++j) {
        v8f acc[4];
#pragma unroll
        for (int g = 0; g < 4; ++g) {
            float init = xg_l[g * Hh + j * 16 + col];  // column-broadcast C init
#pragma unroll
            for (int e = 0; e < 8; ++e) acc[g][e] = init;
            v16bf b0 = load_frag(Wb, g * Hh + j * 16, 0, lane);
            v16bf b1 = load_frag(Wb, g * Hh + j * 16, 32, lane);
            acc[g] = __builtin_amdgcn_wmma_f32_16x16x32_bf16(
                false, a0, false, b0, (short)0, acc[g], false, false);
            acc[g] = __builtin_amdgcn_wmma_f32_16x16x32_bf16(
                false, a1, false, b1, (short)0, acc[g], false, false);
        }
        const int hd = j * 16 + col;
        const float sd = std_l[hd];
#pragma unroll
        for (int r = 0; r < 8; ++r) {
            int p = m0 + r + hi * 8;
            float iv = sigm(acc[0][r]);
            float fv = sigm(acc[1][r]);
            float gv = tanhf(acc[2][r]);
            float ov = sigm(acc[3][r]);
            int cidx = (p * Bb + b) * Hh + hd;
            float cn = fv * c_st[cidx] + iv * gv;
            c_st[cidx] = cn;
            float hl = ov * tanhf(cn);
            unsigned seed = ((unsigned)t * 0x9E3779B9u) ^
                            ((unsigned)(p * Bb + b) * 0x85EBCA6Bu) ^
                            ((unsigned)hd * 0xC2B2AE35u);
            hrep[p * Hh + hd] = hl + gaussian(seed) * sd;
        }
    }
    __syncthreads();

    // ---- per-particle sort projection (Wf, bf) ----
    if (tid < Pp) {
        float s = bff[0];
#pragma unroll
        for (int k = 0; k < Hh; ++k) s += Wf[k] * hrep[tid * Hh + k];
        keys[tid] = s;
        order[tid] = tid;
    }
    __syncthreads();

    // ---- bitonic argsort (ascending) over 256 particles ----
    for (int ksz = 2; ksz <= Pp; ksz <<= 1) {
        for (int jj = ksz >> 1; jj > 0; jj >>= 1) {
            if (tid < Pp) {
                int ixj = tid ^ jj;
                if (ixj > tid) {
                    bool up = ((tid & ksz) == 0);
                    float a = keys[tid], bk = keys[ixj];
                    bool sw = up ? (a > bk) : (a < bk);
                    if (sw) {
                        keys[tid] = bk; keys[ixj] = a;
                        int ta = order[tid]; order[tid] = order[ixj]; order[ixj] = ta;
                    }
                }
            }
            __syncthreads();
        }
    }

    // ---- gather sorted h -> state (bf16), logpdf (Wp, bp), weight incr ----
    if (tid < Pp) {
        int src = order[tid];
        float lp = bp[0];
#pragma unroll
        for (int k = 0; k < Hh; ++k) {
            float hv = hrep[src * Hh + k];
            h_bf[(tid * Bb + b) * Hh + k] = f2bf(hv);
            lp += Wp[k] * hv;
        }
#pragma unroll
        for (int k = 0; k < IN1; ++k) lp += Wp[Hh + k] * xc[k];
        red[tid] = expf(lp);
    }
    __syncthreads();
    for (int off = 128; off > 0; off >>= 1) {
        if (tid < off) red[tid] += red[tid + off];
        __syncthreads();
    }
    if (tid == 0) wacc[b] += red[0] * (1.0f / (float)Pp);
}

// ---------------------------------------------------------------- final output
__global__ __launch_bounds__(64) void final_kernel(
    const unsigned short* __restrict__ h_bf, const float* __restrict__ Wf,
    const float* __restrict__ bff, const float* __restrict__ wacc,
    float* __restrict__ out) {
    __shared__ float part[Hh];
    const int b = blockIdx.x, hd = threadIdx.x;
    float s = 0.0f;
    for (int p = 0; p < Pp; ++p) s += bf2f(h_bf[(p * Bb + b) * Hh + hd]);
    part[hd] = (s * (1.0f / (float)Pp)) * Wf[hd];
    __syncthreads();
    for (int off = 32; off > 0; off >>= 1) {
        if (hd < off) part[hd] += part[hd + off];
        __syncthreads();
    }
    if (hd == 0) {
        out[b]      = part[0] + bff[0];  // y_pred
        out[Bb + b] = wacc[b];           // weights
    }
}

// ---------------------------------------------------------------- launch
extern "C" void kernel_launch(void* const* d_in, const int* in_sizes, int n_in,
                              void* d_out, int out_size, void* d_ws, size_t ws_size,
                              hipStream_t stream) {
    (void)in_sizes; (void)n_in; (void)out_size; (void)ws_size;

    const float* input_data = (const float*)d_in[0];
    const float* y_prev     = (const float*)d_in[1];
    const float* Wv         = (const float*)d_in[2];
    const float* bv         = (const float*)d_in[3];
    const float* W_ih       = (const float*)d_in[4];
    const float* W_hh       = (const float*)d_in[5];
    const float* b_ih       = (const float*)d_in[6];
    const float* b_hh       = (const float*)d_in[7];
    const float* Wp         = (const float*)d_in[8];
    const float* bp         = (const float*)d_in[9];
    const float* Wf         = (const float*)d_in[10];
    const float* bff        = (const float*)d_in[11];
    float* out = (float*)d_out;

    char* ws = (char*)d_ws;
    float*          c_st   = (float*)ws;          ws += sizeof(float) * Pp * Bb * Hh;
    unsigned short* h_bf   = (unsigned short*)ws; ws += sizeof(unsigned short) * Pp * Bb * Hh;
    unsigned short* whh_bf = (unsigned short*)ws; ws += sizeof(unsigned short) * G4 * Hh;
    float*          wacc   = (float*)ws;          ws += sizeof(float) * Bb;

    init_kernel<<<2048, 256, 0, stream>>>(h_bf, c_st, wacc);
    conv_whh_kernel<<<(G4 * Hh + 255) / 256, 256, 0, stream>>>(W_hh, whh_bf);

    const size_t shmem =
        (size_t)Pp * Hh * 2 +       // hA
        (size_t)G4 * Hh * 2 +       // Wb
        (size_t)Pp * Hh * 4 +       // hrep
        (size_t)G4 * 4 +            // xg_l
        (size_t)Hh * 4 +            // std_l
        (size_t)Pp * 4 +            // keys
        (size_t)Pp * 4 +            // order
        (size_t)Pp * 4 +            // red
        16 * 4 +                    // xc
        (size_t)Hh * 4 +            // mean_l
        (size_t)8 * Hh * 4;         // partial

    for (int t = 0; t < Tt; ++t) {
        lstm_step_kernel<<<Bb, 512, shmem, stream>>>(
            input_data, y_prev, Wv, bv, W_ih, b_ih, b_hh, Wp, bp, Wf, bff,
            h_bf, c_st, whh_bf, wacc, t);
    }
    final_kernel<<<Bb, 64, 0, stream>>>(h_bf, Wf, bff, wacc, out);
}
